// MultiheadAttention_cavhe_49340584296501
// MI455X (gfx1250) — matrix operand order repaired and checked by reading, not compiled, not run on hardware
//
#include <hip/hip_runtime.h>

// ---------------------------------------------------------------------------
// MultiheadAttention prefill for MI455X (gfx1250, wave32, WMMA).
// B=4, S=2048, D=1024, H=16, DH=64.
// Pipeline: f32->bf16 convert, fused QKV GEMM (wmma bf16), flash attention
// (wmma bf16 + online softmax), output projection GEMM (wmma bf16, f32 out).
// LDS tiles staged with GLOBAL_LOAD_ASYNC_TO_LDS_B128 (ASYNCcnt) when the
// toolchain exposes the builtin; falls back to synchronous b128 copies.
// ---------------------------------------------------------------------------

typedef __bf16 bf16;
typedef __attribute__((ext_vector_type(16))) __bf16 v16bf;
typedef __attribute__((ext_vector_type(8)))  __bf16 v8bf;
typedef __attribute__((ext_vector_type(8)))  float  v8f;
typedef int vi4 __attribute__((vector_size(16)));  // b128 payload type

static constexpr int Bb = 4;
static constexpr int S  = 2048;
static constexpr int D  = 1024;
static constexpr int H  = 16;
static constexpr int DH = 64;
static constexpr int MROWS = Bb * S;  // 8192

#if __has_builtin(__builtin_amdgcn_global_load_async_to_lds_b128) && \
    __has_builtin(__builtin_amdgcn_s_wait_asynccnt)
#define HAS_ASYNC_LDS 1
#else
#define HAS_ASYNC_LDS 0
#endif

// Copy 16 bytes (8 bf16) global -> LDS.  Async path lands directly in LDS
// (no VGPR round trip), tracked by ASYNCcnt.
__device__ __forceinline__ void copy_b128_g2l(const bf16* g, bf16* l) {
#if HAS_ASYNC_LDS
  __builtin_amdgcn_global_load_async_to_lds_b128(
      (__attribute__((address_space(1))) vi4*)(void*)g,
      (__attribute__((address_space(3))) vi4*)(void*)l, 0, 0);
#else
  *(uint4*)l = *(const uint4*)g;
#endif
}

__device__ __forceinline__ void async_join() {
#if HAS_ASYNC_LDS
  __builtin_amdgcn_s_wait_asynccnt(0);
#endif
}

union Frag { v16bf v; v8bf h[2]; };

// A-fragment (16x32 bf16, lane = M): elements e: k = half*8 + (e&7) + (e>=8)*16
__device__ __forceinline__ v16bf load_frag_a(const bf16* row, int half) {
  Frag u;
  u.h[0] = *(const v8bf*)(row + half * 8);
  u.h[1] = *(const v8bf*)(row + 16 + half * 8);
  return u.v;
}

// B-fragment (32x16 bf16, lane = N): elements e: k = half*8 + (e&7) + (e>=8)*16
__device__ __forceinline__ v16bf load_frag_b(const bf16* row, int half) {
  Frag u;
  u.h[0] = *(const v8bf*)(row + half * 8);
  u.h[1] = *(const v8bf*)(row + 16 + half * 8);
  return u.v;
}

__device__ __forceinline__ v8f wmma_bf16(v16bf a, v16bf b, v8f c) {
  return __builtin_amdgcn_wmma_f32_16x16x32_bf16(false, a, false, b,
                                                 (short)0, c, false, false);
}

// ---------------------------------------------------------------------------
// Stage 0: f32 -> bf16 elementwise convert
// ---------------------------------------------------------------------------
__global__ void convert_f32_bf16(const float* __restrict__ in,
                                 bf16* __restrict__ out, int n) {
  int i = blockIdx.x * blockDim.x + threadIdx.x;
  int stride = gridDim.x * blockDim.x;
  for (; i < n; i += stride) out[i] = (bf16)in[i];
}

// ---------------------------------------------------------------------------
// Stage 1: fused QKV projection.  y = x @ W.T + b   (W is [N,K] row-major)
// grid = (MROWS/128, D/64, 3), block = 256 (8 waves).
// Block tile 128x64, K-step 32.  Wave computes a 32x32 region (2x2 wmma).
// Output written as bf16 in [B, H, S, DH] head layout.
// ---------------------------------------------------------------------------
__global__ void qkv_gemm(const bf16* __restrict__ xb,
                         const bf16* __restrict__ Wq,
                         const bf16* __restrict__ Wk,
                         const bf16* __restrict__ Wv,
                         const float* __restrict__ bq,
                         const float* __restrict__ bk,
                         const float* __restrict__ bv,
                         bf16* __restrict__ outq,
                         bf16* __restrict__ outk,
                         bf16* __restrict__ outv) {
  __shared__ alignas(16) bf16 As[128][32];
  __shared__ alignas(16) bf16 Bs[64][32];

  const bf16* W;
  const float* bias;
  bf16* out;
  if (blockIdx.z == 0)      { W = Wq; bias = bq; out = outq; }
  else if (blockIdx.z == 1) { W = Wk; bias = bk; out = outk; }
  else                      { W = Wv; bias = bv; out = outv; }

  const int lane = threadIdx.x & 31;
  const int wave = threadIdx.x >> 5;
  const int half = lane >> 4;
  const int m0 = blockIdx.x * 128;
  const int n0 = blockIdx.y * 64;
  const int wm = wave >> 1;   // 0..3
  const int wn = wave & 1;    // 0..1

  const int arow = threadIdx.x >> 1, aseg = threadIdx.x & 1;
  const int brow = threadIdx.x >> 2, bseg = threadIdx.x & 3;

  v8f c[2][2] = {};

  for (int k0 = 0; k0 < D; k0 += 32) {
    __syncthreads();  // previous tile fully consumed
    {  // A tile: 128x32 bf16, each thread moves 32B (2 x b128 async)
      const bf16* g = xb + (size_t)(m0 + arow) * D + k0 + aseg * 16;
      bf16* l = &As[arow][aseg * 16];
      copy_b128_g2l(g, l);
      copy_b128_g2l(g + 8, l + 8);
    }
    {  // B tile: 64x32 bf16, each thread moves 16B (1 x b128 async)
      copy_b128_g2l(W + (size_t)(n0 + brow) * D + k0 + bseg * 8,
                    &Bs[brow][bseg * 8]);
    }
    if (k0 + 32 < D) {  // prefetch next K tile (global_prefetch_b8)
      __builtin_prefetch(xb + (size_t)(m0 + arow) * D + k0 + 32, 0, 0);
      __builtin_prefetch(W + (size_t)(n0 + brow) * D + k0 + 32, 0, 0);
    }
    async_join();     // own async copies landed in LDS
    __syncthreads();  // everyone's copies landed

#pragma unroll
    for (int im = 0; im < 2; im++) {
      v16bf a = load_frag_a(&As[wm * 32 + im * 16 + (lane & 15)][0], half);
#pragma unroll
      for (int in = 0; in < 2; in++) {
        v16bf b = load_frag_b(&Bs[wn * 32 + in * 16 + (lane & 15)][0], half);
        c[im][in] = wmma_bf16(a, b, c[im][in]);
      }
    }
  }

  // Epilogue: bias + write to [B,H,S,DH] bf16
#pragma unroll
  for (int im = 0; im < 2; im++) {
#pragma unroll
    for (int in = 0; in < 2; in++) {
      int col = n0 + wn * 32 + in * 16 + (lane & 15);
      int h = col >> 6, dh = col & 63;
      float bb = bias[col];
      int mbase = m0 + wm * 32 + im * 16 + 8 * half;
#pragma unroll
      for (int r = 0; r < 8; r++) {
        int m = mbase + r;
        int b_ = m >> 11;        // / S
        int s_ = m & (S - 1);    // % S
        out[((size_t)(b_ * H + h) * S + s_) * DH + dh] = (bf16)(c[im][in][r] + bb);
      }
    }
  }
}

// ---------------------------------------------------------------------------
// Stage 2: causal flash attention.
// grid = (S/128, B*H), block = 256 (8 waves).  Wave w owns Q rows
// [q0b + 16w, +16).  K/V streamed through LDS in 32-key steps.
// K tile staged async; V tile transposed in-flight (synchronous).
// ---------------------------------------------------------------------------
__global__ void attn_kernel(const bf16* __restrict__ qg,
                            const bf16* __restrict__ kg,
                            const bf16* __restrict__ vg,
                            bf16* __restrict__ ctx) {
  __shared__ alignas(16) bf16 Ks[32][64];   // [key_local][dh]
  __shared__ alignas(16) bf16 Vs[64][32];   // transposed: [dh][key_local]
  __shared__ alignas(16) bf16 Ps[8][16][32];

  const int bh = blockIdx.y;
  const int b_ = bh / H;
  const int h  = bh % H;
  const int lane = threadIdx.x & 31;
  const int wave = threadIdx.x >> 5;
  const int half = lane >> 4;
  const int col  = lane & 15;
  const int q0b  = blockIdx.x * 128;
  const int q0   = q0b + wave * 16;

  const bf16* qbase = qg + (size_t)bh * S * DH;
  const bf16* kbase = kg + (size_t)bh * S * DH;
  const bf16* vbase = vg + (size_t)bh * S * DH;

  // Q fragments for this wave's 16 rows (dh 0..31 and 32..63)
  const bf16* qrow = qbase + (size_t)(q0 + col) * DH;
  v16bf qa0 = load_frag_a(qrow, half);
  v16bf qa1 = load_frag_a(qrow + 32, half);

  v8f o[4] = {};
  float mrow[8], lrow[8];
#pragma unroll
  for (int r = 0; r < 8; r++) { mrow[r] = -1e30f; lrow[r] = 0.0f; }

  const int trow = threadIdx.x >> 3, tseg = threadIdx.x & 7;
  const int ktend = q0b + 128;
  for (int kt = 0; kt < ktend; kt += 32) {
    __syncthreads();
    // K tile: 32x64 bf16, 16B per thread, async into LDS
    copy_b128_g2l(kbase + (size_t)(kt + trow) * DH + tseg * 8,
                  &Ks[trow][tseg * 8]);
    {  // V tile, transposed into LDS (synchronous)
      alignas(16) bf16 tmp[8];
      *(uint4*)tmp = *(const uint4*)(vbase + (size_t)(kt + trow) * DH + tseg * 8);
#pragma unroll
      for (int j = 0; j < 8; j++) Vs[tseg * 8 + j][trow] = tmp[j];
    }
    async_join();
    __syncthreads();

    if (kt < q0 + 16) {  // wave-uniform causal skip
      // ---- scores: S = Q (16x64) x K^T (64x32), two 16x16 C tiles ----
      v8f s0 = {}, s1 = {};
      {
        v16bf b0 = load_frag_b(&Ks[col][0], half);
        v16bf b1 = load_frag_b(&Ks[col][32], half);
        s0 = wmma_bf16(qa0, b0, s0);
        s0 = wmma_bf16(qa1, b1, s0);
        v16bf b2 = load_frag_b(&Ks[16 + col][0], half);
        v16bf b3 = load_frag_b(&Ks[16 + col][32], half);
        s1 = wmma_bf16(qa0, b2, s1);
        s1 = wmma_bf16(qa1, b3, s1);
      }

      // ---- scale + causal mask (scale = 1/sqrt(64)) ----
      float p0e[8], p1e[8];
      const int key0 = kt + col, key1 = kt + 16 + col;
#pragma unroll
      for (int r = 0; r < 8; r++) {
        int q = q0 + r + 8 * half;
        float a = s0[r] * 0.125f;
        float b = s1[r] * 0.125f;
        if (key0 > q) a = -1e30f;
        if (key1 > q) b = -1e30f;
        p0e[r] = a;
        p1e[r] = b;
      }

      // ---- online softmax (row reductions within 16-lane halves) ----
#pragma unroll
      for (int r = 0; r < 8; r++) {
        float mx = fmaxf(p0e[r], p1e[r]);
        for (int off = 1; off < 16; off <<= 1)
          mx = fmaxf(mx, __shfl_xor(mx, off, 32));
        float mn = fmaxf(mrow[r], mx);
        float rescale = __expf(mrow[r] - mn);
        mrow[r] = mn;
        float e0 = __expf(p0e[r] - mn);
        float e1 = __expf(p1e[r] - mn);
        float psum = e0 + e1;
        for (int off = 1; off < 16; off <<= 1)
          psum += __shfl_xor(psum, off, 32);
        lrow[r] = lrow[r] * rescale + psum;
        o[0][r] *= rescale;
        o[1][r] *= rescale;
        o[2][r] *= rescale;
        o[3][r] *= rescale;
        // P (bf16) to per-wave LDS tile (C-layout -> A-layout transpose)
        Ps[wave][r + 8 * half][col]      = (bf16)e0;
        Ps[wave][r + 8 * half][16 + col] = (bf16)e1;
      }
      // same-wave LDS ops are in-order per ISA: no barrier needed here

      // ---- PV: O += P (16x32) x V (32x64) ----
      v16bf pa = load_frag_a(&Ps[wave][col][0], half);
#pragma unroll
      for (int c = 0; c < 4; c++) {
        v16bf vb = load_frag_b(&Vs[16 * c + col][0], half);
        o[c] = wmma_bf16(pa, vb, o[c]);
      }
    }
  }

  // ---- epilogue: normalize and write ctx[b, s, h*64+dh] bf16 ----
  float inv[8];
#pragma unroll
  for (int r = 0; r < 8; r++) inv[r] = 1.0f / lrow[r];
#pragma unroll
  for (int c = 0; c < 4; c++) {
    int dh = 16 * c + col;
#pragma unroll
    for (int r = 0; r < 8; r++) {
      int qrow_g = q0 + r + 8 * half;
      ctx[(size_t)(b_ * S + qrow_g) * D + h * DH + dh] = (bf16)(o[c][r] * inv[r]);
    }
  }
}

// ---------------------------------------------------------------------------
// Stage 3: output projection.  out_f32 = ctx_bf16 @ Wo.T + bo
// grid = (MROWS/128, D/64), block = 256.
// ---------------------------------------------------------------------------
__global__ void out_gemm(const bf16* __restrict__ ab,
                         const bf16* __restrict__ W,
                         const float* __restrict__ bias,
                         float* __restrict__ out) {
  __shared__ alignas(16) bf16 As[128][32];
  __shared__ alignas(16) bf16 Bs[64][32];

  const int lane = threadIdx.x & 31;
  const int wave = threadIdx.x >> 5;
  const int half = lane >> 4;
  const int m0 = blockIdx.x * 128;
  const int n0 = blockIdx.y * 64;
  const int wm = wave >> 1;
  const int wn = wave & 1;

  const int arow = threadIdx.x >> 1, aseg = threadIdx.x & 1;
  const int brow = threadIdx.x >> 2, bseg = threadIdx.x & 3;

  v8f c[2][2] = {};

  for (int k0 = 0; k0 < D; k0 += 32) {
    __syncthreads();
    {
      const bf16* g = ab + (size_t)(m0 + arow) * D + k0 + aseg * 16;
      bf16* l = &As[arow][aseg * 16];
      copy_b128_g2l(g, l);
      copy_b128_g2l(g + 8, l + 8);
    }
    copy_b128_g2l(W + (size_t)(n0 + brow) * D + k0 + bseg * 8,
                  &Bs[brow][bseg * 8]);
    if (k0 + 32 < D) {
      __builtin_prefetch(ab + (size_t)(m0 + arow) * D + k0 + 32, 0, 0);
      __builtin_prefetch(W + (size_t)(n0 + brow) * D + k0 + 32, 0, 0);
    }
    async_join();
    __syncthreads();

#pragma unroll
    for (int im = 0; im < 2; im++) {
      v16bf a = load_frag_a(&As[wm * 32 + im * 16 + (lane & 15)][0], half);
#pragma unroll
      for (int in = 0; in < 2; in++) {
        v16bf b = load_frag_b(&Bs[wn * 32 + in * 16 + (lane & 15)][0], half);
        c[im][in] = wmma_bf16(a, b, c[im][in]);
      }
    }
  }

#pragma unroll
  for (int im = 0; im < 2; im++) {
#pragma unroll
    for (int in = 0; in < 2; in++) {
      int col = n0 + wn * 32 + in * 16 + (lane & 15);
      float bb = bias[col];
      int mbase = m0 + wm * 32 + im * 16 + 8 * half;
#pragma unroll
      for (int r = 0; r < 8; r++) {
        out[(size_t)(mbase + r) * D + col] = c[im][in][r] + bb;
      }
    }
  }
}

// ---------------------------------------------------------------------------
// Host launcher
// ---------------------------------------------------------------------------
extern "C" void kernel_launch(void* const* d_in, const int* in_sizes, int n_in,
                              void* d_out, int out_size, void* d_ws, size_t ws_size,
                              hipStream_t stream) {
  const float* x  = (const float*)d_in[0];
  const float* Wq = (const float*)d_in[1];
  const float* bq = (const float*)d_in[2];
  const float* Wk = (const float*)d_in[3];
  const float* bk = (const float*)d_in[4];
  const float* Wv = (const float*)d_in[5];
  const float* bv = (const float*)d_in[6];
  const float* Wo = (const float*)d_in[7];
  const float* bo = (const float*)d_in[8];
  float* out = (float*)d_out;

  // Workspace carve-up (all bf16): xb, 4 weights, q/k/v head-layout, ctx.
  char* ws = (char*)d_ws;
  const size_t xd_bytes = (size_t)Bb * S * D * sizeof(bf16);   // 16 MB
  const size_t w_bytes  = (size_t)D * D * sizeof(bf16);        // 2 MB
  bf16* xb  = (bf16*)ws;               ws += xd_bytes;
  bf16* Wqb = (bf16*)ws;               ws += w_bytes;
  bf16* Wkb = (bf16*)ws;               ws += w_bytes;
  bf16* Wvb = (bf16*)ws;               ws += w_bytes;
  bf16* Wob = (bf16*)ws;               ws += w_bytes;
  bf16* qb  = (bf16*)ws;               ws += xd_bytes;
  bf16* kb  = (bf16*)ws;               ws += xd_bytes;
  bf16* vb  = (bf16*)ws;               ws += xd_bytes;
  bf16* ctx = (bf16*)ws;               ws += xd_bytes;
  // total ~92 MB

  // Stage 0: converts
  convert_f32_bf16<<<1024, 256, 0, stream>>>(x, xb, Bb * S * D);
  convert_f32_bf16<<<512, 256, 0, stream>>>(Wq, Wqb, D * D);
  convert_f32_bf16<<<512, 256, 0, stream>>>(Wk, Wkb, D * D);
  convert_f32_bf16<<<512, 256, 0, stream>>>(Wv, Wvb, D * D);
  convert_f32_bf16<<<512, 256, 0, stream>>>(Wo, Wob, D * D);

  // Stage 1: fused QKV projection
  qkv_gemm<<<dim3(MROWS / 128, D / 64, 3), 256, 0, stream>>>(
      xb, Wqb, Wkb, Wvb, bq, bk, bv, qb, kb, vb);

  // Stage 2: causal flash attention
  attn_kernel<<<dim3(S / 128, Bb * H), 256, 0, stream>>>(qb, kb, vb, ctx);

  // Stage 3: output projection (f32 out + bias)
  out_gemm<<<dim3(MROWS / 128, D / 64), 256, 0, stream>>>(ctx, Wob, bo, out);
}
